// LSTMMasker_23407571763579
// MI455X (gfx1250) — matrix operand order repaired and checked by reading, not compiled, not run on hardware
//
#include <hip/hip_runtime.h>

// ---------------------------------------------------------------------------
// Types
// ---------------------------------------------------------------------------
typedef __attribute__((ext_vector_type(16))) __bf16 v16bf;
typedef __attribute__((ext_vector_type(8)))  float  v8f;
typedef __attribute__((ext_vector_type(4)))  int    v4i;

#define BB 8        // batch
#define CC 512      // channels
#define TT 1000     // time
#define HH 512      // hidden
#define G4 2048     // 4*H
#define MROWS 8000  // T*B

// CDNA5 async global->LDS copies (ASYNCcnt path), guarded for toolchain support.
#if defined(__AMDGCN__) && \
    __has_builtin(__builtin_amdgcn_global_load_async_to_lds_b128) && \
    __has_builtin(__builtin_amdgcn_s_wait_asynccnt)
#define HAS_ASYNC_LDS 1
typedef __attribute__((address_space(1))) v4i as1_v4i;
typedef __attribute__((address_space(3))) v4i as3_v4i;
#else
#define HAS_ASYNC_LDS 0
#endif

__device__ __forceinline__ __bf16 f2bf(float f) {
    union { float f; unsigned u; } v; v.f = f;
    unsigned r = v.u + 0x7FFFu + ((v.u >> 16) & 1u);   // round-to-nearest-even
    union { unsigned short s; __bf16 b; } o;
    o.s = (unsigned short)(r >> 16);
    return o.b;
}

__device__ __forceinline__ float sigf(float x) {
    return 1.0f / (1.0f + __expf(-x));
}

// A-fragment (16x32 bf16) from a row-major [16, ldk] strip.
// Lane l holds row (l&15); halves 0..7 = K [c0, c0+8), halves 8..15 = K [c0+16, c0+24),
// c0 = (l<16) ? 0 : 8.  (CDNA5 ISA 7.12.2, 16-bit A layout.)
__device__ __forceinline__ v16bf load_a_frag(const __bf16* base, int ldk,
                                             int row, int kk, int c0) {
    union { v16bf v; uint4 u[2]; } a;
    const __bf16* p = base + row * ldk + kk + c0;
    a.u[0] = *(const uint4*)(p);
    a.u[1] = *(const uint4*)(p + 16);
    return a.v;
}

// ---------------------------------------------------------------------------
// 1) GlobLN statistics: one block per batch element
// ---------------------------------------------------------------------------
__global__ void globln_stats_kernel(const float* __restrict__ x,
                                    float* __restrict__ stats) {
    const int b   = blockIdx.x;
    const int tid = threadIdx.x;
    const float* xb = x + (size_t)b * (CC * TT);
    float s = 0.f, q = 0.f;
    for (int i = tid; i < CC * TT; i += blockDim.x) {
        float v = xb[i];
        s += v; q += v * v;
    }
    __shared__ float rs[256];
    __shared__ float rq[256];
    rs[tid] = s; rq[tid] = q;
    __syncthreads();
    for (int off = 128; off > 0; off >>= 1) {
        if (tid < off) { rs[tid] += rs[tid + off]; rq[tid] += rq[tid + off]; }
        __syncthreads();
    }
    if (tid == 0) {
        const float n = (float)(CC * TT);
        float mean = rs[0] / n;
        float var  = rq[0] / n - mean * mean;
        stats[2 * b]     = mean;
        stats[2 * b + 1] = rsqrtf(var + 1e-8f);
    }
}

// 2) Normalize + transpose [B,C,T] -> bf16 [T,B,C]
__global__ void globln_norm_kernel(const float* __restrict__ x,
                                   const float* __restrict__ gamma,
                                   const float* __restrict__ beta,
                                   const float* __restrict__ stats,
                                   __bf16* __restrict__ act0) {
    int idx = blockIdx.x * 256 + threadIdx.x;
    if (idx >= BB * CC * TT) return;
    int b   = idx / (CC * TT);
    int rem = idx - b * (CC * TT);
    int c   = rem / TT;
    int t   = rem - c * TT;
    float v = (x[idx] - stats[2 * b]) * stats[2 * b + 1] * gamma[c] + beta[c];
    act0[((size_t)t * BB + b) * CC + c] = f2bf(v);
}

// 3) Weight convert + transpose: fp32 W[Nrows,K] -> bf16 WT[K,Nrows]
__global__ void weight_transpose_kernel(const float* __restrict__ W,
                                        __bf16* __restrict__ WT,
                                        int Nrows, int K) {
    int idx = blockIdx.x * 256 + threadIdx.x;
    if (idx >= Nrows * K) return;
    int n = idx / K;
    int k = idx - n * K;
    WT[(size_t)k * Nrows + n] = f2bf(W[idx]);
}

// ---------------------------------------------------------------------------
// 4) WMMA GEMM: out[M,N] = A[M,K](bf16) * Bw[K,N](bf16) + bias
//    mode 0: store fp32 to out[m*N+n]           (xproj)
//    mode 1: sigmoid + scatter to out[(b*N+n)*T+t], m = t*B+b   (final mask)
//    Block = 256 threads (8 waves); block tile = 16M x 512N; wave = 4 n-tiles.
// ---------------------------------------------------------------------------
__global__ void wmma_gemm_kernel(const __bf16* __restrict__ A,
                                 const __bf16* __restrict__ Bw,
                                 const float* __restrict__ bias,
                                 float* __restrict__ out,
                                 int N, int K, int mode, int T) {
    __shared__ __bf16 As[16 * 1024];   // 32 KB: one 16-row A strip (K <= 1024)

    const int tid  = threadIdx.x;
    const int wave = tid >> 5;
    const int lane = tid & 31;
    const int m0   = blockIdx.y * 16;

    // Stage A strip [16, K] into LDS (async-to-LDS when available).
    const __bf16* Ag = A + (size_t)m0 * K;
    const int nchunks = (16 * K) >> 3;        // 16B chunks
#if HAS_ASYNC_LDS
    for (int i = tid; i < nchunks; i += 256) {
        __builtin_amdgcn_global_load_async_to_lds_b128(
            (as1_v4i*)(Ag + i * 8), (as3_v4i*)(As + i * 8), 0, 0);
    }
    __builtin_amdgcn_s_wait_asynccnt(0);
#else
    for (int i = tid; i < nchunks; i += 256)
        ((uint4*)As)[i] = ((const uint4*)Ag)[i];
#endif
    __syncthreads();

    const int nwave = blockIdx.x * 512 + wave * 64;   // first col of this wave
    const int row   = lane & 15;
    const int c0    = (lane < 16) ? 0 : 8;

    v8f acc[4] = {v8f{}, v8f{}, v8f{}, v8f{}};

    // B pointer: lane l = column (l&15), K-rows kk+c0+s (s<8) / kk+c0+16+s; strides by 32 rows per k-step.
    const __bf16* bp = Bw + (size_t)c0 * N + nwave + row;
    const size_t  bstep = (size_t)32 * N;

    for (int kk = 0; kk < K; kk += 32) {
        v16bf a = load_a_frag(As, K, row, kk, c0);
        if (kk + 32 < K)
            __builtin_prefetch((const void*)(bp + bstep), 0, 0);
        // Load ALL four B fragments, then do the four WMMAs -> partial waits.
        union { v16bf v; uint4 u[2]; } bu[4];
#pragma unroll
        for (int j = 0; j < 4; ++j) {
#pragma unroll
            for (int s = 0; s < 8; ++s) {
                bu[j].v[s]     = bp[(size_t)s * N + j * 16];
                bu[j].v[8 + s] = bp[(size_t)(16 + s) * N + j * 16];
            }
        }
#pragma unroll
        for (int j = 0; j < 4; ++j) {
            acc[j] = __builtin_amdgcn_wmma_f32_16x16x32_bf16(
                false, a, false, bu[j].v, (short)0, acc[j], false, false);
        }
        bp += bstep;
    }

    const int mrow0 = m0 + ((lane < 16) ? 0 : 8);
    const int ncol  = lane & 15;
#pragma unroll
    for (int j = 0; j < 4; ++j) {
        int n = nwave + j * 16 + ncol;
        float bv = bias[n];
#pragma unroll
        for (int r = 0; r < 8; ++r) {
            int m = mrow0 + r;
            float v = acc[j][r] + bv;
            if (mode == 0) {
                out[(size_t)m * N + n] = v;
            } else {
                int t = m >> 3, b = m & 7;
                out[((size_t)b * N + n) * T + t] = sigf(v);
            }
        }
    }
}

// ---------------------------------------------------------------------------
// 5) Persistent bidirectional LSTM recurrence.
//    grid = 2 (dir), block = 1024 threads = 32 waves.
//    h state [16(pad),512] bf16 in LDS; c state in per-lane registers.
//    Wave w owns columns j in [w*16, w*16+16); its 4 WMMA n-tiles are the
//    i/f/g/o gates for those j (n = w*16 + gate*512) -> no cross-wave mixing.
// ---------------------------------------------------------------------------
__global__ void lstm_recurrent_kernel(const float* __restrict__ xpF,
                                      const float* __restrict__ xpB,
                                      const __bf16* __restrict__ WhF,
                                      const __bf16* __restrict__ WhB,
                                      __bf16* __restrict__ act,   // [T,B,2H]
                                      int T) {
    const int dir = blockIdx.x;
    const float*  xp = dir ? xpB : xpF;
    const __bf16* Wh = dir ? WhB : WhF;

    __shared__ __bf16 h_lds[16 * HH];   // 16 KB, rows 8..15 stay zero (M pad)

    const int tid  = threadIdx.x;
    const int wave = tid >> 5;
    const int lane = tid & 31;

    // zero h (all 16 rows)
    {
        uint4 z; z.x = z.y = z.z = z.w = 0u;
        for (int i = tid; i < (16 * HH * 2) / 16; i += 1024)
            ((uint4*)h_lds)[i] = z;
    }
    float creg[8];
#pragma unroll
    for (int r = 0; r < 8; ++r) creg[r] = 0.f;
    __syncthreads();

    const int row   = lane & 15;
    const int c0    = (lane < 16) ? 0 : 8;
    const int jbase = wave * 16;
    const __bf16* bp0   = Wh + (size_t)c0 * G4 + jbase + row;   // row kk+c0, col jbase+(lane&15)
    const size_t  bstep = (size_t)32 * G4;

    for (int step = 0; step < T; ++step) {
        const int t = dir ? (T - 1 - step) : step;

        v8f acc[4] = {v8f{}, v8f{}, v8f{}, v8f{}};
        const __bf16* bp = bp0;
        for (int kk = 0; kk < HH; kk += 32) {
            v16bf a = load_a_frag(h_lds, HH, row, kk, c0);
            // Load all 4 gate fragments first, then issue the 4 WMMAs.
            union { v16bf v; uint4 u[2]; } bu[4];
#pragma unroll
            for (int g = 0; g < 4; ++g) {
#pragma unroll
                for (int s = 0; s < 8; ++s) {
                    bu[g].v[s]     = bp[(size_t)s * G4 + g * HH];
                    bu[g].v[8 + s] = bp[(size_t)(16 + s) * G4 + g * HH];
                }
            }
#pragma unroll
            for (int g = 0; g < 4; ++g) {
                acc[g] = __builtin_amdgcn_wmma_f32_16x16x32_bf16(
                    false, a, false, bu[g].v, (short)0, acc[g], false, false);
            }
            bp += bstep;
        }

        __syncthreads();   // all h_lds reads for this step done

        if (lane < 16) {
            // lanes 0..15 hold D rows M=r (batch r); columns j = jbase + lane
            const int j = jbase + lane;
            const float* xr = xp + (size_t)t * BB * G4;
#pragma unroll
            for (int r = 0; r < 8; ++r) {
                const size_t ro = (size_t)r * G4;
                float gi = acc[0][r] + xr[ro + j];
                float gf = acc[1][r] + xr[ro + HH + j];
                float gg = acc[2][r] + xr[ro + 2 * HH + j];
                float go = acc[3][r] + xr[ro + 3 * HH + j];
                float cc = sigf(gf) * creg[r] + sigf(gi) * tanhf(gg);
                creg[r]  = cc;
                float hh = sigf(go) * tanhf(cc);
                __bf16 hb = f2bf(hh);
                h_lds[r * HH + j] = hb;
                act[((size_t)(t * BB + r)) * (2 * HH) + dir * HH + j] = hb;
            }
        }
        __syncthreads();   // h updated before next step's matmul
    }
}

// ---------------------------------------------------------------------------
// Host launcher
// ---------------------------------------------------------------------------
extern "C" void kernel_launch(void* const* d_in, const int* in_sizes, int n_in,
                              void* d_out, int out_size, void* d_ws, size_t ws_size,
                              hipStream_t stream) {
    (void)in_sizes; (void)n_in; (void)out_size; (void)ws_size;

    const float* x     = (const float*)d_in[0];
    const float* gamma = (const float*)d_in[1];
    const float* beta  = (const float*)d_in[2];
    const float* Wih[8]; const float* Whh[8]; const float* bg[8];
    for (int i = 0; i < 8; ++i) {
        int base = 3 + i * 3;
        Wih[i] = (const float*)d_in[base];
        Whh[i] = (const float*)d_in[base + 1];
        bg[i]  = (const float*)d_in[base + 2];
    }
    const float* W_out = (const float*)d_in[27];
    const float* b_out = (const float*)d_in[28];
    float* out = (float*)d_out;

    // ---- carve workspace ----
    char* ws = (char*)d_ws;
    size_t off = 0;
    auto alloc = [&](size_t bytes) -> void* {
        void* p = ws + off;
        off = (off + bytes + 255) & ~(size_t)255;
        return p;
    };
    float*  stats = (float*) alloc(2 * BB * sizeof(float));
    __bf16* act0  = (__bf16*)alloc((size_t)TT * BB * CC * 2);
    __bf16* actA  = (__bf16*)alloc((size_t)TT * BB * 2 * HH * 2);
    __bf16* actB  = (__bf16*)alloc((size_t)TT * BB * 2 * HH * 2);
    float*  xpF   = (float*) alloc((size_t)TT * BB * G4 * 4);
    float*  xpB   = (float*) alloc((size_t)TT * BB * G4 * 4);
    __bf16* wihT[8]; __bf16* whhT[8];
    int Kin[8];
    for (int i = 0; i < 8; ++i) {
        Kin[i]  = (i < 2) ? CC : 2 * HH;
        wihT[i] = (__bf16*)alloc((size_t)Kin[i] * G4 * 2);
        whhT[i] = (__bf16*)alloc((size_t)HH * G4 * 2);
    }
    __bf16* woutT = (__bf16*)alloc((size_t)(2 * HH) * (2 * CC) * 2);

    // ---- GlobLN + transpose to [T,B,C] bf16 ----
    globln_stats_kernel<<<BB, 256, 0, stream>>>(x, stats);
    {
        int total = BB * CC * TT;
        globln_norm_kernel<<<(total + 255) / 256, 256, 0, stream>>>(
            x, gamma, beta, stats, act0);
    }

    // ---- weight transpose/convert to bf16 [K,N] ----
    for (int i = 0; i < 8; ++i) {
        int tIh = G4 * Kin[i];
        weight_transpose_kernel<<<(tIh + 255) / 256, 256, 0, stream>>>(
            Wih[i], wihT[i], G4, Kin[i]);
        int tHh = G4 * HH;
        weight_transpose_kernel<<<(tHh + 255) / 256, 256, 0, stream>>>(
            Whh[i], whhT[i], G4, HH);
    }
    {
        int tOut = (2 * CC) * (2 * HH);
        weight_transpose_kernel<<<(tOut + 255) / 256, 256, 0, stream>>>(
            W_out, woutT, 2 * CC, 2 * HH);
    }

    // ---- LSTM stack ----
    const __bf16* in = act0;
    int K = CC;
    __bf16* louts[4] = {actA, actB, actA, actB};
    for (int l = 0; l < 4; ++l) {
        dim3 ggrid(G4 / 512, MROWS / 16);
        wmma_gemm_kernel<<<ggrid, 256, 0, stream>>>(
            in, wihT[2 * l],     bg[2 * l],     xpF, G4, K, 0, TT);
        wmma_gemm_kernel<<<ggrid, 256, 0, stream>>>(
            in, wihT[2 * l + 1], bg[2 * l + 1], xpB, G4, K, 0, TT);
        lstm_recurrent_kernel<<<2, 1024, 0, stream>>>(
            xpF, xpB, whhT[2 * l], whhT[2 * l + 1], louts[l], TT);
        in = louts[l];
        K  = 2 * HH;
    }

    // ---- final projection + sigmoid, scatter to [B, NSRC*C, T] ----
    {
        dim3 ggrid((2 * CC) / 512, MROWS / 16);
        wmma_gemm_kernel<<<ggrid, 256, 0, stream>>>(
            in, woutT, b_out, out, 2 * CC, 2 * HH, 1, TT);
    }
}
